// InstanceLoss_14637248545068
// MI455X (gfx1250) — compile-verified, matrix operand order checked by reading, not run on hardware
//
#include <hip/hip_runtime.h>
#include <hip/hip_bf16.h>

// NT-Xent instance loss, fused flash-logsumexp over sim = z z^T / T.
// N = 16384 rows, D = 128. GEMM in f16 WMMA (16x16x32) with f32 accumulate;
// sim is never materialized. One wave per 16-row block, A fragments resident
// in VGPRs, double-buffered B tiles streamed (z-f16 = 4MB is L2-resident).
// Online logsumexp runs in the exp2 domain using raw v_exp_f32/v_log_f32.

typedef __attribute__((ext_vector_type(16))) _Float16 v16h;
typedef __attribute__((ext_vector_type(8)))  _Float16 v8h;
typedef __attribute__((ext_vector_type(8)))  float    v8f;

#define EMBED_DIM 128
#define LOG2E     1.4426950408889634f
#define SCALE_L2  (2.0f * LOG2E)        // (1/T) * log2(e), T = 0.5
#define LN2       0.6931471805599453f
#define MASK_NEG  (-3.0e38f)

// Raw hardware transcendentals (no OCML ldexp fixup; args are <= 0 here and
// underflow-to-zero is exactly the behavior we want).
#define EXP2F(x)  __builtin_amdgcn_exp2f(x)   // v_exp_f32 (base 2)
#define LOG2F(x)  __builtin_amdgcn_logf(x)    // v_log_f32 (base 2)

// ---------------------------------------------------------------------------
// f32 -> f16 conversion of z = concat(z_i, z_j)
// ---------------------------------------------------------------------------
__global__ __launch_bounds__(256)
void ntxent_convert_f16(const float* __restrict__ zi,
                        const float* __restrict__ zj,
                        _Float16* __restrict__ zh, int total) {
    int i = blockIdx.x * blockDim.x + threadIdx.x;
    if (i < total) {
        zh[i]         = (_Float16)zi[i];
        zh[total + i] = (_Float16)zj[i];
    }
}

// ---------------------------------------------------------------------------
// Fragment loader. CDNA5 wave32 layout for 16-bit A (16x32, MxK):
//   lane L: row = L&15, half = L>>4
//   v16h element e:  K = (e<8) ? half*8 + e : 16 + half*8 + (e-8)
// B (32x16, KxN) mirrors it with N = L&15; B = z^T, so a B-column is a z-row.
// p points at (row base + half*8); chunk c lives at p + 32*c.
// ---------------------------------------------------------------------------
__device__ __forceinline__ v16h load_frag(const _Float16* __restrict__ p) {
    v8h lo = *(const v8h*)(p);        // K = k0 + half*8 + [0..7]
    v8h hi = *(const v8h*)(p + 16);   // K = k0 + 16 + half*8 + [0..7]
    v16h r;
#pragma unroll
    for (int e = 0; e < 8; ++e) { r[e] = lo[e]; r[8 + e] = hi[e]; }
    return r;
}

// Branchless online-logsumexp update in exp2 domain (single v_exp_f32).
__device__ __forceinline__ void online_update(float v, float& m, float& s) {
    float d = m - v;                      // >= 0 when max unchanged
    float e = EXP2F(-fabsf(d));
    float t = (d >= 0.0f) ? 1.0f : e;
    float u = (d >= 0.0f) ? e : 1.0f;
    s = s * t + u;
    m = fmaxf(m, v);
}

// ---------------------------------------------------------------------------
// Main fused kernel: one wave handles rows [row0, row0+16).
// ---------------------------------------------------------------------------
__global__ __launch_bounds__(256)
void ntxent_wmma_kernel(const _Float16* __restrict__ zh,
                        float* __restrict__ rowloss, int n /* = TRAIN_NUM */) {
    const int N    = 2 * n;
    const int lane = threadIdx.x & 31;
    const int Mh   = lane & 15;           // fragment row/col within tile
    const int half = lane >> 4;           // K-half selector
    const int rb   = blockIdx.x * (blockDim.x >> 5) + (threadIdx.x >> 5);
    const int row0 = rb * 16;
    if (row0 >= N) return;                // wave-uniform: EXEC stays all-ones

    // A fragments: this wave's 16 rows, K = 0..127 (resident all kernel)
    const _Float16* arow = zh + (size_t)(row0 + Mh) * EMBED_DIM + half * 8;
    const v16h a0 = load_frag(arow);
    const v16h a1 = load_frag(arow + 32);
    const v16h a2 = load_frag(arow + 64);
    const v16h a3 = load_frag(arow + 96);

    // Online state: C-layout row for VGPR r is row0 + r + 8*half, col = Mh.
    float m[8], s[8], pos[8];
#pragma unroll
    for (int r = 0; r < 8; ++r) { m[r] = -1.0e30f; s[r] = 0.0f; pos[r] = 0.0f; }

    // Diagonal & positive-pair columns each live in ONE wave-uniform tile
    // (row0 and n are multiples of 16).
    const int jtDiag = row0 >> 4;
    const int jtPos  = ((row0 < n) ? (row0 + n) : (row0 - n)) >> 4;

    const int ntiles = N >> 4;
    const _Float16* cbase = zh + (size_t)Mh * EMBED_DIM + half * 8;
    auto colptr = [&](int jt) { return cbase + (size_t)(jt << 4) * EMBED_DIM; };

    auto process_tile = [&](int jt, const v16h& B0, const v16h& B1,
                            const v16h& B2, const v16h& B3) {
        v8f c = {};
        c = __builtin_amdgcn_wmma_f32_16x16x32_f16(false, a0, false, B0, (short)0, c, false, false);
        c = __builtin_amdgcn_wmma_f32_16x16x32_f16(false, a1, false, B1, (short)0, c, false, false);
        c = __builtin_amdgcn_wmma_f32_16x16x32_f16(false, a2, false, B2, (short)0, c, false, false);
        c = __builtin_amdgcn_wmma_f32_16x16x32_f16(false, a3, false, B3, (short)0, c, false, false);

        if (jt == jtDiag || jt == jtPos) {          // rare special tiles
            const int col = (jt << 4) + Mh;
#pragma unroll
            for (int r = 0; r < 8; ++r) {
                const int rowg = row0 + (half << 3) + r;
                const int pc   = (rowg < n) ? rowg + n : rowg - n;
                float v = c[r] * SCALE_L2;
                v = (col == rowg) ? MASK_NEG : v;   // mask diagonal
                pos[r] = (col == pc) ? v : pos[r];  // capture positive
                online_update(v, m[r], s[r]);
            }
        } else {                                    // fast path
#pragma unroll
            for (int r = 0; r < 8; ++r)
                online_update(c[r] * SCALE_L2, m[r], s[r]);
        }
    };

    // Software pipeline: double-buffered B fragments, 2 tiles per iteration.
    v16h b0, b1, b2, b3, d0, d1, d2, d3;
    { const _Float16* p = colptr(0);
      b0 = load_frag(p); b1 = load_frag(p + 32);
      b2 = load_frag(p + 64); b3 = load_frag(p + 96); }

    for (int jt = 0; jt < ntiles; jt += 2) {
        { const _Float16* p = colptr(jt + 1);       // loads for tile jt+1
          d0 = load_frag(p); d1 = load_frag(p + 32);
          d2 = load_frag(p + 64); d3 = load_frag(p + 96); }
        __builtin_prefetch(colptr((jt + 3 < ntiles) ? jt + 3 : 0), 0, 1);

        process_tile(jt, b0, b1, b2, b3);

        { const int j2 = (jt + 2 < ntiles) ? jt + 2 : 0;
          const _Float16* p = colptr(j2);           // loads for tile jt+2
          b0 = load_frag(p); b1 = load_frag(p + 32);
          b2 = load_frag(p + 64); b3 = load_frag(p + 96); }

        process_tile(jt + 1, d0, d1, d2, d3);
    }

    // Merge 16 per-lane partials per row (butterfly within each 16-lane half)
#pragma unroll
    for (int r = 0; r < 8; ++r) {
        float mm = m[r], ss = s[r], pp = pos[r];
#pragma unroll
        for (int off = 1; off < 16; off <<= 1) {
            float m2 = __shfl_xor(mm, off, 32);
            float s2 = __shfl_xor(ss, off, 32);
            float d  = mm - m2;
            float e  = EXP2F(-fabsf(d));
            float t  = (d >= 0.0f) ? 1.0f : e;
            float u  = (d >= 0.0f) ? e : 1.0f;
            ss = ss * t + s2 * u;
            mm = fmaxf(mm, m2);
            pp += __shfl_xor(pp, off, 32);
        }
        if (Mh == 0) {
            const int rowg = row0 + (half << 3) + r;
            rowloss[rowg] = (mm + LOG2F(ss) - pp) * LN2;   // (lse - pos), nats
        }
    }
}

// ---------------------------------------------------------------------------
// Deterministic final reduction (single block, fixed tree order).
// ---------------------------------------------------------------------------
__global__ __launch_bounds__(256)
void ntxent_reduce_kernel(const float* __restrict__ rowloss,
                          float* __restrict__ out, int N) {
    __shared__ float sdata[256];
    float acc = 0.0f;
    for (int i = threadIdx.x; i < N; i += 256) acc += rowloss[i];
    sdata[threadIdx.x] = acc;
    __syncthreads();
    for (int stride = 128; stride > 0; stride >>= 1) {
        if ((int)threadIdx.x < stride)
            sdata[threadIdx.x] += sdata[threadIdx.x + stride];
        __syncthreads();
    }
    if (threadIdx.x == 0) out[0] = sdata[0] / (float)N;
}

// ---------------------------------------------------------------------------
extern "C" void kernel_launch(void* const* d_in, const int* in_sizes, int n_in,
                              void* d_out, int out_size, void* d_ws, size_t ws_size,
                              hipStream_t stream) {
    const float* zi = (const float*)d_in[0];
    const float* zj = (const float*)d_in[1];
    const int total = in_sizes[0];          // n * 128
    const int n     = total / EMBED_DIM;    // 8192
    const int N     = 2 * n;                // 16384

    // workspace: [ z_f16 : N*128 halves ][ rowloss : N floats ]
    _Float16* zh      = (_Float16*)d_ws;
    float*    rowloss = (float*)((char*)d_ws + (size_t)N * EMBED_DIM * sizeof(_Float16));

    ntxent_convert_f16<<<(total + 255) / 256, 256, 0, stream>>>(zi, zj, zh, total);

    const int waves  = N / 16;              // 1024 row blocks
    const int blocks = (waves + 7) / 8;     // 8 waves (256 threads) per block
    ntxent_wmma_kernel<<<blocks, 256, 0, stream>>>(zh, rowloss, n);

    ntxent_reduce_kernel<<<1, 256, 0, stream>>>(rowloss, (float*)d_out, N);
}
